// BayesianLinearEnsembleLayer_1632087572792
// MI455X (gfx1250) — matrix operand order repaired and checked by reading, not compiled, not run on hardware
//
#include <hip/hip_runtime.h>
#include <hip/hip_bf16.h>

// ---------------------------------------------------------------------------
// BayesianLinearEnsembleLayer for MI455X (gfx1250, wave32, WMMA)
//   pass 1: w = mu + softplus(rho)*eps  -> bf16, stored TRANSPOSED wT[M][O][I]
//   pass 2: bias = bmu + softplus(brho)*beps (f32)
//   pass 3: out[m] = x[m] (f32->bf16) @ wT[m]^T + bias  via v_wmma_f32_16x16x32_bf16
//           128x256 block tile, 8 waves (2x4), 64x64 wave tile (4x4 WMMA frags),
//           double-buffered LDS; B tile filled by GLOBAL_LOAD_ASYNC_TO_LDS_B128
//           (ASYNCcnt DMA) when available, A tile register-staged (needs f32->bf16).
// ---------------------------------------------------------------------------

typedef __bf16 bf16_t;
typedef __attribute__((ext_vector_type(16))) __bf16 v16bf;
typedef __attribute__((ext_vector_type(8)))  __bf16 v8bf;
typedef __attribute__((ext_vector_type(4)))  __bf16 v4bf;
typedef __attribute__((ext_vector_type(8)))  float   v8f;
typedef int v4i_b128 __attribute__((vector_size(16)));   // matches builtin param type

#define M_ENS 8
#define BATCH 4096
#define IN_F  2048
#define OUT_F 2048

#define BM 128
#define BN 256
#define BK 32
#define KPAD 8          // LDS row stride = 40 bf16 = 80 bytes (16B aligned, bank-friendly)
#define NSTEP (IN_F / BK)

#if defined(__has_builtin)
#if __has_builtin(__builtin_amdgcn_global_load_async_to_lds_b128) && \
    __has_builtin(__builtin_amdgcn_s_wait_asynccnt)
#define USE_ASYNC_B 1
#endif
#endif
#ifndef USE_ASYNC_B
#define USE_ASYNC_B 0
#endif

#if USE_ASYNC_B
#define AS_LDS __attribute__((address_space(3)))
#endif

__device__ __forceinline__ float softplusf(float x) {
    return fmaxf(x, 0.0f) + log1pf(__expf(-fabsf(x)));
}

// ---------------------------------------------------------------------------
// Pass 1: generate bf16 weights, transposed to [M][O][I] via LDS tile
// grid (O/32, I/32, M), block (32, 8)
// ---------------------------------------------------------------------------
__global__ __launch_bounds__(256) void gen_w_transpose(
    const float* __restrict__ mu, const float* __restrict__ rho,
    const float* __restrict__ eps, bf16_t* __restrict__ wT)
{
    __shared__ bf16_t tile[32][33];
    const int m  = blockIdx.z;
    const int i0 = blockIdx.y * 32;
    const int o0 = blockIdx.x * 32;
    const int tx = threadIdx.x;          // 0..31 (fast dim -> coalesced)
    const int ty = threadIdx.y;          // 0..7

    const size_t base = (size_t)m * IN_F * OUT_F;
#pragma unroll
    for (int r = 0; r < 4; ++r) {
        const int i = i0 + ty + 8 * r;
        const size_t idx = base + (size_t)i * OUT_F + (size_t)(o0 + tx);
        const float w = mu[idx] + softplusf(rho[idx]) * eps[idx];
        tile[ty + 8 * r][tx] = (bf16_t)w;
    }
    __syncthreads();
    const size_t baseT = (size_t)m * OUT_F * IN_F;
#pragma unroll
    for (int r = 0; r < 4; ++r) {
        const int o = o0 + ty + 8 * r;
        wT[baseT + (size_t)o * IN_F + (size_t)(i0 + tx)] = tile[tx][ty + 8 * r];
    }
}

// ---------------------------------------------------------------------------
// Pass 2: bias (f32), M*O = 16384 elements
// ---------------------------------------------------------------------------
__global__ __launch_bounds__(256) void gen_bias(
    const float* __restrict__ bmu, const float* __restrict__ brho,
    const float* __restrict__ beps, float* __restrict__ bias)
{
    const int idx = blockIdx.x * blockDim.x + threadIdx.x;
    if (idx < M_ENS * OUT_F)
        bias[idx] = bmu[idx] + softplusf(brho[idx]) * beps[idx];
}

// ---------------------------------------------------------------------------
// Pass 3: per-member GEMM: Y[m] = X[m] @ W[m] + bias[m]
// grid (OUT_F/BN, BATCH/BM, M), 256 threads = 8 waves (2 row-groups x 4 col-groups)
// ---------------------------------------------------------------------------
__global__ __launch_bounds__(256) void bayes_gemm(
    const float* __restrict__ X, const bf16_t* __restrict__ Wt,
    const float* __restrict__ bias, float* __restrict__ Y)
{
    __shared__ bf16_t As[2][BM][BK + KPAD];   // [buf][row m][k]
    __shared__ bf16_t Bs[2][BN][BK + KPAD];   // [buf][col n][k]

    const int m       = blockIdx.z;
    const int rowBase = blockIdx.y * BM;
    const int colBase = blockIdx.x * BN;

    const int tid  = threadIdx.x;
    const int lane = tid & 31;
    const int wave = tid >> 5;       // 0..7
    const int wm   = wave >> 2;      // 0..1 : 64-row group
    const int wn   = wave & 3;       // 0..3 : 64-col group
    const int l15  = lane & 15;
    const int hi   = lane >> 4;      // lane-half selector (K-half per ISA layout)

    const float*  A  = X  + (size_t)m * BATCH * IN_F + (size_t)rowBase * IN_F;
    const bf16_t* Bt = Wt + (size_t)m * OUT_F * IN_F + (size_t)colBase * IN_F;

    // global->LDS mappings
    const int a_cg  = tid & 7;       // k chunk of 4 floats
    const int a_row = tid >> 3;      // 0..31 (+32*r)
    const int b_cg  = tid & 3;       // k chunk of 8 bf16
    const int b_row = tid >> 2;      // 0..63 (+64*r)

    v8f acc[4][4] = {};
    float4 aReg[4];
#if !USE_ASYNC_B
    v8bf   bReg[4];
#endif

    // ---- prologue: stage tile 0 into buffer 0
#pragma unroll
    for (int r = 0; r < 4; ++r)
        aReg[r] = *(const float4*)(A + (size_t)(a_row + 32 * r) * IN_F + a_cg * 4);
#if USE_ASYNC_B
#pragma unroll
    for (int r = 0; r < 4; ++r)
        __builtin_amdgcn_global_load_async_to_lds_b128(
            (v4i_b128*)(Bt + (size_t)(b_row + 64 * r) * IN_F + b_cg * 8),
            (AS_LDS v4i_b128*)(AS_LDS void*)&Bs[0][b_row + 64 * r][b_cg * 8],
            0, 0);
#else
#pragma unroll
    for (int r = 0; r < 4; ++r)
        bReg[r] = *(const v8bf*)(Bt + (size_t)(b_row + 64 * r) * IN_F + b_cg * 8);
#endif
#pragma unroll
    for (int r = 0; r < 4; ++r) {
        v4bf p;
        p[0] = (bf16_t)aReg[r].x; p[1] = (bf16_t)aReg[r].y;
        p[2] = (bf16_t)aReg[r].z; p[3] = (bf16_t)aReg[r].w;
        *(v4bf*)&As[0][a_row + 32 * r][a_cg * 4] = p;
    }
#if USE_ASYNC_B
    __builtin_amdgcn_s_wait_asynccnt(0);
#else
#pragma unroll
    for (int r = 0; r < 4; ++r)
        *(v8bf*)&Bs[0][b_row + 64 * r][b_cg * 8] = bReg[r];
#endif
    __syncthreads();

    for (int ks = 0; ks < NSTEP; ++ks) {
        const int cur = ks & 1;
        const int nxt = cur ^ 1;
        const int k0n = (ks + 1) * BK;

        // ---- prefetch next tile (B: async DMA straight to LDS; A: via registers)
        if (ks + 1 < NSTEP) {
#if USE_ASYNC_B
#pragma unroll
            for (int r = 0; r < 4; ++r)
                __builtin_amdgcn_global_load_async_to_lds_b128(
                    (v4i_b128*)(Bt + (size_t)(b_row + 64 * r) * IN_F + (k0n + b_cg * 8)),
                    (AS_LDS v4i_b128*)(AS_LDS void*)&Bs[nxt][b_row + 64 * r][b_cg * 8],
                    0, 0);
#else
#pragma unroll
            for (int r = 0; r < 4; ++r)
                bReg[r] = *(const v8bf*)(Bt + (size_t)(b_row + 64 * r) * IN_F +
                                         (k0n + b_cg * 8));
#endif
#pragma unroll
            for (int r = 0; r < 4; ++r)
                aReg[r] = *(const float4*)(A + (size_t)(a_row + 32 * r) * IN_F +
                                           (k0n + a_cg * 4));
        }

        // ---- fragments from LDS buffer `cur`
        // A layout: el j -> K = hi*8+j (j<8) / 16+hi*8+(j-8); row = lane&15
        // B layout: lane = col n, el j -> K = hi*8+j (j<8) / 16+hi*8+(j-8)
        v16bf af[4], bfr[4];
#pragma unroll
        for (int im = 0; im < 4; ++im) {
            const int row = wm * 64 + im * 16 + l15;
            const v8bf lo = *(const v8bf*)&As[cur][row][hi * 8];
            const v8bf h8 = *(const v8bf*)&As[cur][row][16 + hi * 8];
#pragma unroll
            for (int j = 0; j < 8; ++j) { af[im][j] = lo[j]; af[im][8 + j] = h8[j]; }
        }
#pragma unroll
        for (int in = 0; in < 4; ++in) {
            const int n = wn * 64 + in * 16 + l15;
            const v8bf lo = *(const v8bf*)&Bs[cur][n][hi * 8];
            const v8bf h8 = *(const v8bf*)&Bs[cur][n][16 + hi * 8];
#pragma unroll
            for (int j = 0; j < 8; ++j) { bfr[in][j] = lo[j]; bfr[in][8 + j] = h8[j]; }
        }

        // ---- 16 WMMAs per wave per k-step (4x4 of 16x16x32)
#pragma unroll
        for (int im = 0; im < 4; ++im)
#pragma unroll
            for (int in = 0; in < 4; ++in)
                acc[im][in] = __builtin_amdgcn_wmma_f32_16x16x32_bf16(
                    false, af[im], false, bfr[in],
                    (short)0, acc[im][in], false, false);

        // ---- stage next A tile into the other buffer
        if (ks + 1 < NSTEP) {
#pragma unroll
            for (int r = 0; r < 4; ++r) {
                v4bf p;
                p[0] = (bf16_t)aReg[r].x; p[1] = (bf16_t)aReg[r].y;
                p[2] = (bf16_t)aReg[r].z; p[3] = (bf16_t)aReg[r].w;
                *(v4bf*)&As[nxt][a_row + 32 * r][a_cg * 4] = p;
            }
#if !USE_ASYNC_B
#pragma unroll
            for (int r = 0; r < 4; ++r)
                *(v8bf*)&Bs[nxt][b_row + 64 * r][b_cg * 8] = bReg[r];
#endif
        }
#if USE_ASYNC_B
        __builtin_amdgcn_s_wait_asynccnt(0);   // B tile for next iter landed in LDS
#endif
        __syncthreads();
    }

    // ---- epilogue: C/D layout el r -> row = hi*8+r, col = lane&15; add bias
    float* C = Y + (size_t)m * BATCH * OUT_F + (size_t)rowBase * OUT_F + colBase;
#pragma unroll
    for (int in = 0; in < 4; ++in) {
        const int col = wn * 64 + in * 16 + l15;
        const float bv = bias[m * OUT_F + colBase + col];
#pragma unroll
        for (int im = 0; im < 4; ++im) {
            const int rbase = wm * 64 + im * 16 + hi * 8;
#pragma unroll
            for (int r = 0; r < 8; ++r)
                C[(size_t)(rbase + r) * OUT_F + col] = acc[im][in][r] + bv;
        }
    }
}

// ---------------------------------------------------------------------------
extern "C" void kernel_launch(void* const* d_in, const int* in_sizes, int n_in,
                              void* d_out, int out_size, void* d_ws, size_t ws_size,
                              hipStream_t stream)
{
    const float* x    = (const float*)d_in[0];
    const float* wmu  = (const float*)d_in[1];
    const float* wrho = (const float*)d_in[2];
    const float* bmu  = (const float*)d_in[3];
    const float* brho = (const float*)d_in[4];
    const float* ew   = (const float*)d_in[5];
    const float* eb   = (const float*)d_in[6];
    float* out = (float*)d_out;

    // workspace: [ wT bf16 : M*I*O ][ bias f32 : M*O ]
    bf16_t* wT    = (bf16_t*)d_ws;
    float*  biasb = (float*)((char*)d_ws + (size_t)M_ENS * IN_F * OUT_F * sizeof(bf16_t));

    gen_w_transpose<<<dim3(OUT_F / 32, IN_F / 32, M_ENS), dim3(32, 8), 0, stream>>>(
        wmu, wrho, ew, wT);
    gen_bias<<<(M_ENS * OUT_F + 255) / 256, 256, 0, stream>>>(bmu, brho, eb, biasb);
    bayes_gemm<<<dim3(OUT_F / BN, BATCH / BM, M_ENS), 256, 0, stream>>>(
        x, wT, biasb, out);

    (void)in_sizes; (void)n_in; (void)out_size; (void)ws_size;
}
